// MolGANDiscriminator_87960930222583
// MI455X (gfx1250) — matrix operand order repaired and checked by reading, not compiled, not run on hardware
//
#include <hip/hip_runtime.h>

// ---------------------------------------------------------------------------
// MolGAN discriminator on MI455X (gfx1250, wave32, WMMA).
// Pipeline:
//   Abf  = bf16(A)                       [B][N][K=2048]  (k = m*4 + r)
//   Xbf  = bf16(X); Wt_l = bf16(W_l^T)   [r][o][f]
//   for l in 1..3:
//     XWt = proj(H_{l-1}, Wt_l)          [B][O][K]  (pre-transposed for agg)
//     H_l = relu(agg(Abf, XWt) + b_l)    [B][N][O]  bf16
//   g = sum_n H_3 ; out = (leakyrelu(g@mw1+mb1)) @ mw2 + mb2
// GEMMs: v_wmma_f32_16x16x32_bf16, fp32 accumulate.
// agg_kernel stages the block-shared A tile in LDS via
// global_load_async_to_lds_b128 (ASYNCcnt), double buffered.
// ---------------------------------------------------------------------------

typedef __attribute__((ext_vector_type(16))) __bf16 bf16x16;
typedef __attribute__((ext_vector_type(8)))  __bf16 bf16x8;
typedef __attribute__((ext_vector_type(4)))  __bf16 bf16x4;
typedef __attribute__((ext_vector_type(8)))  float  f32x8;

#define B_SZ   32
#define N_SZ   512
#define R_SZ   4
#define F_IN   64
#define HID    256
#define K_AGG  (N_SZ * R_SZ)   // 2048

// agg LDS staging geometry
#define CHUNK_K   128                 // k elements staged per buffer
#define KP        (CHUNK_K + 8)       // padded row: 272B -> conflict-free rows
#define NROWS     32                  // n rows per block
#define BUF_ELEMS (NROWS * KP)        // 4352 bf16 = 8704 B per buffer

// ---- WMMA fragment loaders (layouts per CDNA5 ISA 7.12.2, wave32) ----------

// A-matrix 16x32 bf16, source row-major [M][K] with leading dim ld (global).
// lanes 0-15: row M=lane, K {0..7,16..23}; lanes 16-31: row M=lane-16, K {8..15,24..31}
__device__ __forceinline__ bf16x16 load_frag_a(const __bf16* __restrict__ base,
                                               int ld, int lane) {
  const __bf16* row = base + (size_t)(lane & 15) * ld;
  const int k0 = (lane & 16) ? 8 : 0;
  union { bf16x16 v; bf16x8 h[2]; } u;
  u.h[0] = *(const bf16x8*)(row + k0);
  u.h[1] = *(const bf16x8*)(row + k0 + 16);
  return u.v;
}

// Same A fragment but from the padded LDS tile (row stride KP elements).
__device__ __forceinline__ bf16x16 load_frag_a_lds(const __bf16* base, int lane) {
  const __bf16* row = base + (size_t)(lane & 15) * KP + ((lane & 16) ? 8 : 0);
  union { bf16x16 v; bf16x8 h[2]; } u;
  u.h[0] = *(const bf16x8*)(row);        // ds_load_b128
  u.h[1] = *(const bf16x8*)(row + 16);   // ds_load_b128
  return u.v;
}

// B-matrix 32x16 bf16 from TRANSPOSED storage [N][K], leading dim ld.
__device__ __forceinline__ bf16x16 load_frag_bt(const __bf16* __restrict__ base,
                                                int ld, int lane) {
  const __bf16* col = base + (size_t)(lane & 15) * ld + ((lane & 16) ? 8 : 0);
  union { bf16x16 v; bf16x8 h[2]; } u;
  u.h[0] = *(const bf16x8*)(col);
  u.h[1] = *(const bf16x8*)(col + 16);
  return u.v;
}

__device__ __forceinline__ f32x8 wmma_bf16(bf16x16 a, bf16x16 b, f32x8 c) {
  return __builtin_amdgcn_wmma_f32_16x16x32_bf16(false, a, false, b,
                                                 (short)0, c, false, false);
}

// gfx1250 async copy: 16B global -> LDS, tracked by ASYNCcnt.
__device__ __forceinline__ void async_copy16(unsigned lds_addr, const void* gaddr) {
  asm volatile("global_load_async_to_lds_b128 %0, %1, off"
               :: "v"(lds_addr), "v"(gaddr) : "memory");
}

// ---- fp32 -> bf16 bulk convert (vectorized x4) -----------------------------
__global__ __launch_bounds__(256) void cvt_f32_bf16_v4(
    const float4* __restrict__ in, bf16x4* __restrict__ out, int n4) {
  int i = blockIdx.x * 256 + threadIdx.x;
  if (i < n4) {
    float4 v = in[i];
    bf16x4 o;
    o[0] = (__bf16)v.x; o[1] = (__bf16)v.y; o[2] = (__bf16)v.z; o[3] = (__bf16)v.w;
    out[i] = o;
  }
}

// ---- W[r][f][o] fp32 -> Wt[r][o][f] bf16 (tiny) ----------------------------
__global__ __launch_bounds__(256) void wtrans_kernel(
    const float* __restrict__ W, __bf16* __restrict__ Wt, int Fin) {
  int idx = blockIdx.x * 256 + threadIdx.x;
  int total = R_SZ * HID * Fin;
  if (idx >= total) return;
  int f = idx % Fin;
  int o = (idx / Fin) % HID;
  int r = idx / (Fin * HID);
  Wt[idx] = (__bf16)W[((size_t)r * Fin + f) * HID + o];
}

// ---- projection: XWt[b][o][m*4+r] = sum_f Hin[b][m][f] * Wt[r][o][f] -------
// One wave per block: 16(o) x 16(m) tile, all 4 relations at once.
__global__ __launch_bounds__(32) void proj_kernel(
    const __bf16* __restrict__ Hin, const __bf16* __restrict__ Wt,
    __bf16* __restrict__ XWt, int Fin) {
  const int lane = threadIdx.x;
  const int mt = blockIdx.x;   // 0..31  (m tile)
  const int ot = blockIdx.y;   // 0..15  (o tile)
  const int b  = blockIdx.z;   // 0..31

  const __bf16* hbase = Hin + ((size_t)b * N_SZ + mt * 16) * Fin;
  f32x8 acc0 = {}, acc1 = {}, acc2 = {}, acc3 = {};

  for (int f0 = 0; f0 < Fin; f0 += 32) {
    bf16x16 bfrag = load_frag_bt(hbase + f0, Fin, lane);   // shared across r
    const __bf16* w0 = Wt + ((size_t)0 * HID + ot * 16) * Fin + f0;
    const __bf16* w1 = Wt + ((size_t)1 * HID + ot * 16) * Fin + f0;
    const __bf16* w2 = Wt + ((size_t)2 * HID + ot * 16) * Fin + f0;
    const __bf16* w3 = Wt + ((size_t)3 * HID + ot * 16) * Fin + f0;
    acc0 = wmma_bf16(load_frag_a(w0, Fin, lane), bfrag, acc0);
    acc1 = wmma_bf16(load_frag_a(w1, Fin, lane), bfrag, acc1);
    acc2 = wmma_bf16(load_frag_a(w2, Fin, lane), bfrag, acc2);
    acc3 = wmma_bf16(load_frag_a(w3, Fin, lane), bfrag, acc3);
  }

  // D layout: VGPR j, lanes 0-15 -> M=j (o), N=lane (m); lanes 16-31 -> M=j+8.
  const int m = mt * 16 + (lane & 15);
  const int orow0 = ot * 16 + ((lane & 16) ? 8 : 0);
  __bf16* outb = XWt + (size_t)b * HID * K_AGG + (size_t)m * R_SZ;
#pragma unroll
  for (int j = 0; j < 8; ++j) {
    bf16x4 v;
    v[0] = (__bf16)acc0[j]; v[1] = (__bf16)acc1[j];
    v[2] = (__bf16)acc2[j]; v[3] = (__bf16)acc3[j];
    *(bf16x4*)(outb + (size_t)(orow0 + j) * K_AGG) = v;   // 8B coalesced pack
  }
}

// ---- aggregation: H[b][n][o] = relu(sum_k Abf[b][n][k]*XWt[b][o][k] + bias[o])
// Block = 8 waves over 32(n) x 256(o). Shared 32x128 A tile staged in LDS via
// async copies (double buffered); wave w owns o-range [32w, 32w+32).
__global__ __launch_bounds__(256) void agg_kernel(
    const __bf16* __restrict__ Abf, const __bf16* __restrict__ XWt,
    const float* __restrict__ bias, __bf16* __restrict__ Hout) {
  __shared__ alignas(16) __bf16 smem[2 * BUF_ELEMS];

  const int tid  = threadIdx.x;
  const int lane = tid & 31;
  const int wave = tid >> 5;     // 0..7
  const int n0 = blockIdx.x * 32;
  const int b  = blockIdx.y;
  const int o0 = wave * 32;

  const __bf16* aBase = Abf + ((size_t)b * N_SZ + n0) * K_AGG;
  const __bf16* bB0 = XWt + ((size_t)b * HID + o0) * K_AGG;
  const __bf16* bB1 = bB0 + (size_t)16 * K_AGG;

  const unsigned sbase = (unsigned)(size_t)(void*)smem;   // LDS byte offset
  const int srow = tid >> 4;     // 0..15  (staging row, +16 for 2nd chunk)
  const int scol = tid & 15;     // 16B column within staged row

  // Stage A[n0..n0+31][k0..k0+127] into buffer `bsel` (512 x 16B, 2/thread).
  auto stage = [&](int bsel, int k0) {
    const unsigned l0 = sbase + (unsigned)bsel * (BUF_ELEMS * 2)
                      + (unsigned)srow * (KP * 2) + (unsigned)scol * 16;
    const __bf16* g0 = aBase + (size_t)srow * K_AGG + k0 + scol * 8;
    async_copy16(l0, g0);
    async_copy16(l0 + 16u * (KP * 2), g0 + (size_t)16 * K_AGG);
  };

  f32x8 acc00 = {}, acc01 = {}, acc10 = {}, acc11 = {};

  stage(0, 0);
  const int NCH = K_AGG / CHUNK_K;   // 16
  for (int kc = 0; kc < NCH; ++kc) {
    const int buf = kc & 1;
    const int k0 = kc * CHUNK_K;
    if (kc + 1 < NCH) {
      stage(buf ^ 1, k0 + CHUNK_K);                  // overlap next DMA
      __builtin_prefetch(bB0 + k0 + CHUNK_K, 0, 3);  // global_prefetch_b8
      __builtin_prefetch(bB1 + k0 + CHUNK_K, 0, 3);
      asm volatile("s_wait_asynccnt 0x2" ::: "memory");  // current buf landed
    } else {
      asm volatile("s_wait_asynccnt 0x0" ::: "memory");
    }
    __syncthreads();   // all waves' copies visible

    const __bf16* lb = smem + buf * BUF_ELEMS;
#pragma unroll
    for (int sub = 0; sub < CHUNK_K; sub += 32) {
      bf16x16 a0 = load_frag_a_lds(lb + sub, lane);            // rows 0-15
      bf16x16 a1 = load_frag_a_lds(lb + 16 * KP + sub, lane);  // rows 16-31
      bf16x16 f0 = load_frag_bt(bB0 + k0 + sub, K_AGG, lane);
      bf16x16 f1 = load_frag_bt(bB1 + k0 + sub, K_AGG, lane);
      acc00 = wmma_bf16(a0, f0, acc00);
      acc01 = wmma_bf16(a0, f1, acc01);
      acc10 = wmma_bf16(a1, f0, acc10);
      acc11 = wmma_bf16(a1, f1, acc11);
    }
    __syncthreads();   // readers done before buf is overwritten next iter
  }

  const float bo0 = bias[o0 + (lane & 15)];
  const float bo1 = bias[o0 + 16 + (lane & 15)];
  __bf16* hb = Hout + (size_t)b * N_SZ * HID;
  const int nrow = (lane & 16) ? 8 : 0;
  const int oA = o0 + (lane & 15);
  const int oB = oA + 16;
#pragma unroll
  for (int j = 0; j < 8; ++j) {
    const int nA = n0 + nrow + j;
    const int nB = nA + 16;
    hb[(size_t)nA * HID + oA] = (__bf16)fmaxf(acc00[j] + bo0, 0.f);
    hb[(size_t)nA * HID + oB] = (__bf16)fmaxf(acc01[j] + bo1, 0.f);
    hb[(size_t)nB * HID + oA] = (__bf16)fmaxf(acc10[j] + bo0, 0.f);
    hb[(size_t)nB * HID + oB] = (__bf16)fmaxf(acc11[j] + bo1, 0.f);
  }
}

// ---- graph readout: g[b][o] = sum_n H[b][n][o] -----------------------------
__global__ __launch_bounds__(256) void readout_kernel(
    const __bf16* __restrict__ H, float* __restrict__ g) {
  const int b = blockIdx.x;
  const int o = threadIdx.x;
  const __bf16* p = H + (size_t)b * N_SZ * HID + o;
  float s = 0.f;
  for (int n = 0; n < N_SZ; ++n) s += (float)p[(size_t)n * HID];
  g[b * HID + o] = s;
}

// ---- MLP head: out[b] = leakyrelu(g@mw1+mb1, 0.2) @ mw2 + mb2 --------------
__global__ __launch_bounds__(256) void mlp_kernel(
    const float* __restrict__ g, const float* __restrict__ mw1,
    const float* __restrict__ mb1, const float* __restrict__ mw2,
    const float* __restrict__ mb2, float* __restrict__ out) {
  __shared__ float gs[B_SZ * HID];
  __shared__ float acc[B_SZ];
  const int t = threadIdx.x;
  for (int i = t; i < B_SZ * HID; i += 256) gs[i] = g[i];
  if (t < B_SZ) acc[t] = 0.f;
  __syncthreads();

  const float w2 = mw2[t];
  const float bias1 = mb1[t];
  for (int b = 0; b < B_SZ; ++b) {
    float s = bias1;
    for (int i = 0; i < HID; ++i) s += gs[b * HID + i] * mw1[i * HID + t];
    s = (s > 0.f) ? s : 0.2f * s;
    atomicAdd(&acc[b], s * w2);   // ds_add_f32
  }
  __syncthreads();
  if (t < B_SZ) out[t] = acc[t] + mb2[0];
}

// ---------------------------------------------------------------------------
extern "C" void kernel_launch(void* const* d_in, const int* in_sizes, int n_in,
                              void* d_out, int out_size, void* d_ws, size_t ws_size,
                              hipStream_t stream) {
  const float* A   = (const float*)d_in[0];
  const float* X   = (const float*)d_in[1];
  const float* W1  = (const float*)d_in[2];
  const float* b1  = (const float*)d_in[3];
  const float* W2  = (const float*)d_in[4];
  const float* b2  = (const float*)d_in[5];
  const float* W3  = (const float*)d_in[6];
  const float* b3  = (const float*)d_in[7];
  const float* mw1 = (const float*)d_in[8];
  const float* mb1 = (const float*)d_in[9];
  const float* mw2 = (const float*)d_in[10];
  const float* mb2 = (const float*)d_in[11];
  float* out = (float*)d_out;

  // workspace carve-up (256B aligned)
  char* ws = (char*)d_ws;
  size_t off = 0;
  auto carve = [&](size_t bytes) -> void* {
    void* p = ws + off;
    off += (bytes + 255) & ~(size_t)255;
    return p;
  };
  const size_t nA  = (size_t)B_SZ * N_SZ * N_SZ * R_SZ;   // 33,554,432
  const size_t nX  = (size_t)B_SZ * N_SZ * F_IN;          //  1,048,576
  const size_t nH  = (size_t)B_SZ * N_SZ * HID;           //  4,194,304
  const size_t nXW = (size_t)B_SZ * HID * K_AGG;          // 16,777,216

  __bf16* Abf = (__bf16*)carve(nA * 2);
  __bf16* XWt = (__bf16*)carve(nXW * 2);
  __bf16* Ha  = (__bf16*)carve(nH * 2);
  __bf16* Hb  = (__bf16*)carve(nH * 2);
  __bf16* Xbf = (__bf16*)carve(nX * 2);
  __bf16* W1t = (__bf16*)carve((size_t)R_SZ * HID * F_IN * 2);
  __bf16* W2t = (__bf16*)carve((size_t)R_SZ * HID * HID * 2);
  __bf16* W3t = (__bf16*)carve((size_t)R_SZ * HID * HID * 2);
  float*  g   = (float*)carve((size_t)B_SZ * HID * 4);

  // 1) one-time bf16 conversions (A stays L2-resident at 67 MB thereafter)
  {
    int n4 = (int)(nA / 4);
    cvt_f32_bf16_v4<<<(n4 + 255) / 256, 256, 0, stream>>>(
        (const float4*)A, (bf16x4*)Abf, n4);
  }
  {
    int n4 = (int)(nX / 4);
    cvt_f32_bf16_v4<<<(n4 + 255) / 256, 256, 0, stream>>>(
        (const float4*)X, (bf16x4*)Xbf, n4);
  }
  wtrans_kernel<<<(R_SZ * HID * F_IN + 255) / 256, 256, 0, stream>>>(W1, W1t, F_IN);
  wtrans_kernel<<<(R_SZ * HID * HID + 255) / 256, 256, 0, stream>>>(W2, W2t, HID);
  wtrans_kernel<<<(R_SZ * HID * HID + 255) / 256, 256, 0, stream>>>(W3, W3t, HID);

  const dim3 projGrid(N_SZ / 16, HID / 16, B_SZ);   // (32, 16, 32)
  const dim3 aggGrid(N_SZ / 32, B_SZ);              // (16, 32)

  // 2) layer 1
  proj_kernel<<<projGrid, 32, 0, stream>>>(Xbf, W1t, XWt, F_IN);
  agg_kernel<<<aggGrid, 256, 0, stream>>>(Abf, XWt, b1, Ha);
  // 3) layer 2
  proj_kernel<<<projGrid, 32, 0, stream>>>(Ha, W2t, XWt, HID);
  agg_kernel<<<aggGrid, 256, 0, stream>>>(Abf, XWt, b2, Hb);
  // 4) layer 3
  proj_kernel<<<projGrid, 32, 0, stream>>>(Hb, W3t, XWt, HID);
  agg_kernel<<<aggGrid, 256, 0, stream>>>(Abf, XWt, b3, Ha);

  // 5) readout + MLP head
  readout_kernel<<<B_SZ, HID, 0, stream>>>(Ha, g);
  mlp_kernel<<<1, 256, 0, stream>>>(g, mw1, mb1, mw2, mb2, out);
}
